// APLoss_31464930410736
// MI455X (gfx1250) — matrix operand order repaired and checked by reading, not compiled, not run on hardware
//
#include <hip/hip_runtime.h>

typedef __attribute__((ext_vector_type(16))) _Float16 v16h;
typedef __attribute__((ext_vector_type(8)))  _Float16 v8h;
typedef __attribute__((ext_vector_type(8)))  float    v8f;

#define BB 2
#define CC 128
#define HH 32
#define WW 32
#define NN 1024   // H*W queries per batch (= M database size)
#define NQ 25

// --- Kernel 1: transpose desc1 (B,C,N) -> (B,N,C) f16 rows (WMMA-A friendly) ---
__global__ __launch_bounds__(CC) void k_convert_q(const float* __restrict__ desc1,
                                                  _Float16* __restrict__ qh) {
    int bm = blockIdx.x;          // b*NN + n
    int c  = threadIdx.x;
    int b  = bm >> 10;
    int n  = bm & (NN - 1);
    float v = desc1[((size_t)(b * CC + c)) * NN + n];
    qh[(size_t)bm * CC + c] = (_Float16)v;
}

// --- Kernel 2: bilinear grid-sample desc2 -> (B,M,C) f16 rows (WMMA-B friendly) ---
__global__ __launch_bounds__(CC) void k_sample(const float* __restrict__ desc2,
                                               const float* __restrict__ gridp,
                                               _Float16* __restrict__ dbh) {
    int bm = blockIdx.x;          // b*NN + m  (m = h*W + w, matches grid layout)
    int c  = threadIdx.x;
    int b  = bm >> 10;
    float gxn = gridp[(size_t)bm * 2 + 0];
    float gyn = gridp[(size_t)bm * 2 + 1];
    float gx = ((gxn + 1.0f) * (float)WW - 1.0f) * 0.5f;
    float gy = ((gyn + 1.0f) * (float)HH - 1.0f) * 0.5f;
    float x0f = floorf(gx), y0f = floorf(gy);
    float wx1 = gx - x0f, wy1 = gy - y0f;
    int x0 = (int)x0f, y0 = (int)y0f;
    const float* base = desc2 + ((size_t)(b * CC + c)) * NN;
    float acc = 0.0f;
#pragma unroll
    for (int dy = 0; dy < 2; ++dy) {
#pragma unroll
        for (int dx = 0; dx < 2; ++dx) {
            int xi = x0 + dx, yi = y0 + dy;
            float w = (dy ? wy1 : 1.0f - wy1) * (dx ? wx1 : 1.0f - wx1);
            if (xi >= 0 && xi < WW && yi >= 0 && yi < HH)
                acc += w * base[yi * WW + xi];
        }
    }
    dbh[(size_t)bm * CC + c] = (_Float16)acc;
}

// --- Kernel 3: fused WMMA GEMM + 2-bin soft-histogram AP ---
// grid = B*64 workgroups (16-query tile each), block = 128 (4 waves).
__global__ __launch_bounds__(128) void k_qap(const _Float16* __restrict__ qh,
                                             const _Float16* __restrict__ dbh,
                                             const int* __restrict__ label,
                                             float* __restrict__ apout) {
    __shared__ float s_nbs[16 * NQ];
    __shared__ float s_rec[16 * NQ];
    int tid = threadIdx.x;
    for (int i = tid; i < 16 * NQ; i += 128) { s_nbs[i] = 0.0f; s_rec[i] = 0.0f; }
    __syncthreads();

    int b     = blockIdx.x >> 6;
    int ntile = blockIdx.x & 63;
    int wave  = tid >> 5;
    int lane  = tid & 31;
    int hl    = lane >> 4;   // half-wave select
    int ll    = lane & 15;

    // A fragments: 16x32 f16 per K-step; lane row = ll, K = hl*16 .. hl*16+15 (contiguous halves).
    const _Float16* qrow = qh + ((size_t)(b * NN + ntile * 16 + ll)) * CC + hl * 16;
    v16h afrag[4];
#pragma unroll
    for (int ks = 0; ks < 4; ++ks) {
        v8h lo = *(const v8h*)(qrow + ks * 32);
        v8h hi = *(const v8h*)(qrow + ks * 32 + 8);
        afrag[ks] = __builtin_shufflevector(lo, hi, 0,1,2,3,4,5,6,7,8,9,10,11,12,13,14,15);
    }

    for (int it = 0; it < 16; ++it) {
        int mt = wave + it * 4;                 // m-tile handled by this wave
        // B fragment: 32x16 f16; lane col = ll, K = hl*16 .. hl*16+15 (contiguous halves)
        const _Float16* drow = dbh + ((size_t)(b * NN + mt * 16 + ll)) * CC + hl * 16;
        v8f acc = {};
#pragma unroll
        for (int ks = 0; ks < 4; ++ks) {
            v16h bfrag = *(const v16h*)(drow + ks * 32);
            acc = __builtin_amdgcn_wmma_f32_16x16x32_f16(
                false, afrag[ks], false, bfrag, (short)0, acc, false, false);
        }
        int m = mt * 16 + ll;
#pragma unroll
        for (int v = 0; v < 8; ++v) {
            int qr = v + hl * 8;                // D row: query offset in tile
            int n  = ntile * 16 + qr;
            int lab = label[((size_t)b * NN + n) * NN + m];
            float s = acc[v];
            // t = (1-s)*a ; triangular kernel hits bins j0 and j0+1 only
            float t = (1.0f - s) * (float)(NQ - 1);
            t = fminf(fmaxf(t, 0.0f), 2.0f * (float)(NQ - 1));
            int j0 = (int)t;
            if (j0 > NQ - 1) j0 = NQ - 1;
            float w0 = fmaxf(0.0f, 1.0f - fabsf(t - (float)j0));
            atomicAdd(&s_nbs[qr * NQ + j0], w0);
            if (lab) atomicAdd(&s_rec[qr * NQ + j0], w0);
            if (j0 < NQ - 1) {
                float w1 = fmaxf(0.0f, 1.0f - fabsf(t - (float)(j0 + 1)));
                if (w1 > 0.0f) {
                    atomicAdd(&s_nbs[qr * NQ + j0 + 1], w1);
                    if (lab) atomicAdd(&s_rec[qr * NQ + j0 + 1], w1);
                }
            }
        }
    }
    __syncthreads();

    // Finalize: cumsum over bins, precision, AP per query (16 queries -> 16 threads)
    if (tid < 16) {
        float tot = 0.0f;
#pragma unroll
        for (int q = 0; q < NQ; ++q) tot += s_rec[tid * NQ + q];
        float cum_n = 0.0f, cum_r = 0.0f, acc_pr = 0.0f;
#pragma unroll
        for (int q = 0; q < NQ; ++q) {
            float r = s_rec[tid * NQ + q];
            cum_n += s_nbs[tid * NQ + q];
            cum_r += r;
            acc_pr += (cum_r / (1e-16f + cum_n)) * r;
        }
        apout[b * NN + ntile * 16 + tid] = acc_pr / tot;
    }
}

// --- Kernel 4: mean over B*N AP values -> scalar ---
__global__ __launch_bounds__(256) void k_reduce(const float* __restrict__ ap,
                                                float* __restrict__ out) {
    __shared__ float sm[256];
    int tid = threadIdx.x;
    float s = 0.0f;
    for (int i = tid; i < BB * NN; i += 256) s += ap[i];
    sm[tid] = s;
    __syncthreads();
    for (int off = 128; off > 0; off >>= 1) {
        if (tid < off) sm[tid] += sm[tid + off];
        __syncthreads();
    }
    if (tid == 0) out[0] = sm[0] * (1.0f / (float)(BB * NN));
}

extern "C" void kernel_launch(void* const* d_in, const int* in_sizes, int n_in,
                              void* d_out, int out_size, void* d_ws, size_t ws_size,
                              hipStream_t stream) {
    const float* desc1 = (const float*)d_in[0];
    const float* desc2 = (const float*)d_in[1];
    // d_in[2] = reliability: unused by the reference computation
    const float* gridp = (const float*)d_in[3];
    const int*   label = (const int*)d_in[4];

    char* ws = (char*)d_ws;
    const size_t QB = (size_t)BB * NN * CC * 2;   // f16 bytes per descriptor set (512 KB)
    _Float16* qh  = (_Float16*)ws;                // (B,N,C) f16
    _Float16* dbh = (_Float16*)(ws + QB);         // (B,M,C) f16 (warped)
    float*    ap  = (float*)(ws + 2 * QB);        // (B*N) f32

    k_convert_q<<<BB * NN, CC, 0, stream>>>(desc1, qh);
    k_sample   <<<BB * NN, CC, 0, stream>>>(desc2, gridp, dbh);
    k_qap      <<<BB * 64, 128, 0, stream>>>(qh, dbh, label, ap);
    k_reduce   <<<1, 256, 0, stream>>>(ap, (float*)d_out);
}
